// PostProcess_69097433858783
// MI455X (gfx1250) — compile-verified, matrix operand order checked by reading, not run on hardware
//
#include <hip/hip_runtime.h>

#ifndef __has_builtin
#define __has_builtin(x) 0
#endif

#define BDIM    256
#define NBIN    4096
#define NSORT   2048
#define KCAP    1024     // padded NMS capacity (pow2 >= MAX_BOX_PRE_NMS)
#define K_PRE   1000     // MAX_BOX_PRE_NMS
#define NS2     1024
#define IOU_THR 0.5f

typedef __attribute__((ext_vector_type(2)))  float    v2f;
typedef __attribute__((ext_vector_type(8)))  float    v8f;
typedef __attribute__((ext_vector_type(16))) _Float16 v16h;

// ---------------- gfx1250-specific helpers ----------------
#if __has_builtin(__builtin_amdgcn_global_load_async_to_lds_b32)
#define HAVE_ASYNC_LDS 1
__device__ __forceinline__ void async_gather_f32(const float* g, float* l) {
  __builtin_amdgcn_global_load_async_to_lds_b32(
      (__attribute__((address_space(1))) int*)g,
      (__attribute__((address_space(3))) int*)l, 0, 0);
}
#endif

#if __has_builtin(__builtin_amdgcn_s_wait_asynccnt)
#define WAIT_ASYNC(n) __builtin_amdgcn_s_wait_asynccnt(n)
#else
#define WAIT_ASYNC(n) asm volatile("s_wait_asynccnt " #n ::: "memory")
#endif

__device__ __forceinline__ unsigned ballot32(bool p) {
#if __has_builtin(__builtin_amdgcn_ballot_w32)
  return __builtin_amdgcn_ballot_w32(p);
#else
  return (unsigned)__ballot((int)p);
#endif
}

__device__ __forceinline__ float neg_inf() { return -__builtin_inff(); }

// descending bitonic sort of (score, id) pairs held in LDS
__device__ inline void bitonic_desc(float* s, int* id, int n) {
  const int tid = threadIdx.x;
  for (unsigned k = 2; k <= (unsigned)n; k <<= 1) {
    for (unsigned j = k >> 1; j > 0; j >>= 1) {
      __syncthreads();
      for (unsigned i = (unsigned)tid; i < (unsigned)n; i += BDIM) {
        unsigned ixj = i ^ j;
        if (ixj > i) {
          float a = s[i], b = s[ixj];
          bool desc = ((i & k) == 0u);
          if (desc ? (a < b) : (a > b)) {
            s[i] = b; s[ixj] = a;
            int t = id[i]; id[i] = id[ixj]; id[ixj] = t;
          }
        }
      }
    }
  }
  __syncthreads();
}

// ---------------- stage-1 dynamic LDS layout (bytes) ----------------
#define S1_ABUF  0                       // 2*BDIM f32   (async double buffer)
#define S1_HIST  (S1_ABUF + 2*BDIM*4)    // NBIN u32
#define S1_SCAL  (S1_HIST + NBIN*4)      // 4 u32: cnt, cutoffbin, validcnt
#define S1_SSC   (S1_SCAL + 16)          // NSORT f32
#define S1_SID   (S1_SSC + NSORT*4)      // NSORT i32
#define S1_BX    (S1_SID + NSORT*4)      // KCAP*4 f32 boxes
#define S1_AREA  (S1_BX + KCAP*16)       // KCAP f32
#define S1_SUP   (S1_AREA + KCAP*4)      // KCAP*32 u32 suppression bitmatrix
#define S1_KEEP  (S1_SUP + KCAP*128)     // 32 u32
#define S1_BYTES (S1_KEEP + 128)

#define S2_HIST  0
#define S2_SCAL  (NBIN*4)
#define S2_SSC   (S2_SCAL + 16)
#define S2_SID   (S2_SSC + NS2*4)
#define S2_BYTES (S2_SID + NS2*4)

// ===================== Stage 1: per-(b,c) top-K + greedy NMS =====================
__global__ void __launch_bounds__(BDIM)
pp_stage1(const float* __restrict__ cls, const float* __restrict__ ta,
          const float* __restrict__ thr_p, int B, int A, int C,
          float* __restrict__ ws_scores, float* __restrict__ ws_boxes) {
  extern __shared__ char smem[];
  float*    abuf  = (float*)(smem + S1_ABUF);
  unsigned* hist  = (unsigned*)(smem + S1_HIST);
  unsigned* scal  = (unsigned*)(smem + S1_SCAL);
  float*    ssc   = (float*)(smem + S1_SSC);
  int*      sid   = (int*)(smem + S1_SID);
  float*    bxs   = (float*)(smem + S1_BX);
  float*    area  = (float*)(smem + S1_AREA);
  unsigned* sup   = (unsigned*)(smem + S1_SUP);
  unsigned* keepw = (unsigned*)(smem + S1_KEEP);

  const int tid = threadIdx.x;
  const int bc  = blockIdx.x;
  const int b   = bc / C, c = bc % C;
  const float th = thr_p[0];
  const float bscale = (float)NBIN / fmaxf(1.0f - th, 1e-6f);
  auto binOf = [&](float s) -> int {
    int v = (int)((s - th) * bscale);
    return v < 0 ? 0 : (v >= NBIN ? NBIN - 1 : v);
  };

  for (int i = tid; i < NBIN; i += BDIM) hist[i] = 0u;
  if (tid < 4) scal[tid] = 0u;
  __syncthreads();

  const float* base = cls + (size_t)b * A * C + c;   // element a: base[a*C]
  const int nIter = (A + BDIM - 1) / BDIM;

  // ---- pass 1: histogram of above-threshold scores (stream via async->LDS) ----
#ifdef HAVE_ASYNC_LDS
  {
    size_t aa = (tid < A) ? (size_t)tid : (size_t)(A - 1);
    async_gather_f32(base + aa * C, &abuf[tid]);
  }
  for (int it = 0; it < nIter; ++it) {
    const int cur = it & 1;
    if (it + 1 < nIter) {
      int an = (it + 1) * BDIM + tid;
      size_t aa = (an < A) ? (size_t)an : (size_t)(A - 1);
      async_gather_f32(base + aa * C, &abuf[(cur ^ 1) * BDIM + tid]);
      WAIT_ASYNC(1);                 // oldest chunk (in-order) has landed in LDS
    } else {
      WAIT_ASYNC(0);
    }
    int a = it * BDIM + tid;
    float s = abuf[cur * BDIM + tid];
    if (a < A && s > th) atomicAdd(&hist[binOf(s)], 1u);
  }
#else
  for (int it = 0; it < nIter; ++it) {
    int a = it * BDIM + tid;
    if (a < A) {
      __builtin_prefetch((const void*)(base + (size_t)(a + BDIM) * C), 0, 0);
      float s = base[(size_t)a * C];
      if (s > th) atomicAdd(&hist[binOf(s)], 1u);
    }
  }
#endif
  __syncthreads();

  // ---- exact rank-K cut from histogram ----
  if (tid == 0) {
    unsigned acc = 0; int cb = 0;
    for (int v = NBIN - 1; v >= 0; --v) {
      acc += hist[v];
      if (acc >= (unsigned)K_PRE) { cb = v; break; }
    }
    scal[1] = (unsigned)cb;
    scal[0] = 0u;
  }
  __syncthreads();
  const int cb = (int)scal[1];

  for (int i = tid; i < NSORT; i += BDIM) { ssc[i] = neg_inf(); sid[i] = -1; }
  __syncthreads();

  // ---- pass 2 (L2-resident): compact survivors ----
  for (int it = 0; it < nIter; ++it) {
    int a = it * BDIM + tid;
    if (a < A) {
      float s = base[(size_t)a * C];
      if (s > th && binOf(s) >= cb) {
        unsigned pos = atomicAdd(&scal[0], 1u);
        if (pos < NSORT) { ssc[pos] = s; sid[pos] = a; }
      }
    }
  }
  __syncthreads();
  if (tid == 0) {
    unsigned m = scal[0];
    if (m > NSORT) m = NSORT;
    if (m > K_PRE) m = K_PRE;
    scal[2] = m;
  }
  bitonic_desc(ssc, sid, NSORT);       // begins with __syncthreads
  const int vc = (int)scal[2];

  // ---- gather boxes, areas; zero suppression matrix ----
  for (int k = tid; k < KCAP; k += BDIM) {
    float x1 = 0.f, y1 = 0.f, x2 = 0.f, y2 = 0.f;
    if (k < vc) {
      const float* tp = ta + ((size_t)b * A + sid[k]) * 4;
      x1 = tp[0]; y1 = tp[1]; x2 = tp[2]; y2 = tp[3];
    }
    bxs[k * 4 + 0] = x1; bxs[k * 4 + 1] = y1;
    bxs[k * 4 + 2] = x2; bxs[k * 4 + 3] = y2;
    area[k] = (x2 - x1) * (y2 - y1);
  }
  for (int i = tid; i < KCAP * 32; i += BDIM) sup[i] = 0u;
  __syncthreads();

  // ---- suppression matrix: 16x16 tiles; areaSum via WMMA, inter via VALU ----
  const int lane = tid & 31;
  const int wid  = tid >> 5;
  const int hi   = lane >> 4;          // 0: lanes 0-15, 1: lanes 16-31
  const int lj   = lane & 15;
  const int tileRows = (vc + 15) >> 4;

  for (int t = wid; t < 64 * 64; t += 8) {
    int ti = t >> 6, tj = t & 63;
    if (tj < ti || ti >= tileRows || tj * 16 >= vc) continue;   // wave-uniform
    int i0 = ti * 16, j0 = tj * 16;

    v8f acc = {};
#if __has_builtin(__builtin_amdgcn_wmma_f32_16x16x4_f32)
    // A[M][k] = {area_i, 1, 0, 0}; B[k][N] = {1, area_j, 0, 0}^T  =>  D = a_i + a_j
    int ii = hi ? 0 : (i0 + lane);
    v2f a2, b2;
    a2[0] = hi ? 0.0f : area[ii];  a2[1] = hi ? 0.0f : 1.0f;
    b2[0] = hi ? 0.0f : 1.0f;      b2[1] = hi ? 0.0f : area[j0 + lj];
    acc = __builtin_amdgcn_wmma_f32_16x16x4_f32(false, a2, false, b2,
                                                (short)0, acc, false, false);
#else
    int ii = hi ? 0 : (i0 + lane);
    v16h ah = {}, bh = {};
    ah[0] = hi ? (_Float16)0 : (_Float16)area[ii];
    ah[1] = hi ? (_Float16)0 : (_Float16)1.0f;
    bh[0] = hi ? (_Float16)0 : (_Float16)1.0f;
    bh[1] = hi ? (_Float16)0 : (_Float16)area[j0 + lj];
    acc = __builtin_amdgcn_wmma_f32_16x16x32_f16(false, ah, false, bh,
                                                 (short)0, acc, false, false);
#endif
    int j = j0 + lj;
    float xj1 = bxs[j * 4 + 0], yj1 = bxs[j * 4 + 1];
    float xj2 = bxs[j * 4 + 2], yj2 = bxs[j * 4 + 3];
#pragma unroll
    for (int r = 0; r < 8; ++r) {
      int i = i0 + r + hi * 8;                  // C/D layout: lanes>=16 hold M=r+8
      float xi1 = bxs[i * 4 + 0], yi1 = bxs[i * 4 + 1];
      float xi2 = bxs[i * 4 + 2], yi2 = bxs[i * 4 + 3];
      float iw = fminf(xi2, xj2) - fmaxf(xi1, xj1);
      float ih = fminf(yi2, yj2) - fmaxf(yi1, yj1);
      float inter = fmaxf(iw, 0.0f) * fmaxf(ih, 0.0f);
      bool s = (j > i) && (i < vc) && (j < vc) &&
               (inter * (1.0f + IOU_THR) > IOU_THR * acc[r]);
      unsigned m = ballot32(s);                 // bits0-15: row i0+r, bits16-31: row i0+r+8
      if (lane == 0) {
        unsigned sh = (unsigned)(j0 & 31);      // 0 or 16
        atomicOr(&sup[(i0 + r)     * 32 + (j0 >> 5)], (m & 0xffffu) << sh);
        atomicOr(&sup[(i0 + r + 8) * 32 + (j0 >> 5)], (m >> 16)     << sh);
      }
    }
  }
  __syncthreads();

  // ---- sequential greedy scan (one wave32; lane L owns keep bits [32L,32L+32)) ----
  if (tid < 32) {
    int bb = tid * 32;
    unsigned kw = (vc >= bb + 32) ? 0xffffffffu
                : (vc <= bb) ? 0u : ((1u << (vc - bb)) - 1u);
    for (int i = 0; i < vc; ++i) {
      unsigned w = __shfl(kw, i >> 5, 32);
      if ((w >> (i & 31)) & 1u) kw &= ~sup[i * 32 + tid];
    }
    keepw[tid] = kw;
  }
  __syncthreads();

  // ---- emit per-(b,c) kept candidates ----
  size_t obase = (size_t)bc * KCAP;
  for (int k = tid; k < KCAP; k += BDIM) {
    bool kept = (k < vc) && ((keepw[k >> 5] >> (k & 31)) & 1u);
    ws_scores[obase + k] = kept ? ssc[k] : neg_inf();
    ws_boxes[(obase + k) * 4 + 0] = bxs[k * 4 + 0];
    ws_boxes[(obase + k) * 4 + 1] = bxs[k * 4 + 1];
    ws_boxes[(obase + k) * 4 + 2] = bxs[k * 4 + 2];
    ws_boxes[(obase + k) * 4 + 3] = bxs[k * 4 + 3];
  }
}

// ===================== Stage 2: per-image top-max_box =====================
__global__ void __launch_bounds__(BDIM)
pp_stage2(const float* __restrict__ ws_scores, const float* __restrict__ ws_boxes,
          const float* __restrict__ thr_p, int B, int C, int maxbox,
          float* __restrict__ out) {
  extern __shared__ char smem[];
  unsigned* hist = (unsigned*)(smem + S2_HIST);
  unsigned* scal = (unsigned*)(smem + S2_SCAL);
  float*    ssc  = (float*)(smem + S2_SSC);
  int*      sid  = (int*)(smem + S2_SID);

  const int tid = threadIdx.x;
  const int b   = blockIdx.x;
  const float th = thr_p[0];
  const float bscale = (float)NBIN / fmaxf(1.0f - th, 1e-6f);
  auto binOf = [&](float s) -> int {
    int v = (int)((s - th) * bscale);
    return v < 0 ? 0 : (v >= NBIN ? NBIN - 1 : v);
  };

  const int N2 = C * KCAP;
  const float* sc = ws_scores + (size_t)b * N2;

  for (int i = tid; i < NBIN; i += BDIM) hist[i] = 0u;
  if (tid < 4) scal[tid] = 0u;
  __syncthreads();

  for (int t = tid; t < N2; t += BDIM) {
    float s = sc[t];
    if (s > th) atomicAdd(&hist[binOf(s)], 1u);
  }
  __syncthreads();
  if (tid == 0) {
    unsigned acc = 0; int cb = 0;
    for (int v = NBIN - 1; v >= 0; --v) {
      acc += hist[v];
      if (acc >= (unsigned)maxbox) { cb = v; break; }
    }
    scal[1] = (unsigned)cb; scal[0] = 0u;
  }
  __syncthreads();
  const int cb = (int)scal[1];

  for (int i = tid; i < NS2; i += BDIM) { ssc[i] = neg_inf(); sid[i] = -1; }
  __syncthreads();
  for (int t = tid; t < N2; t += BDIM) {
    float s = sc[t];
    if (s > th && binOf(s) >= cb) {
      unsigned pos = atomicAdd(&scal[0], 1u);
      if (pos < NS2) { ssc[pos] = s; sid[pos] = t; }
    }
  }
  __syncthreads();
  if (tid == 0) {
    unsigned m = scal[0];
    scal[2] = (m > NS2) ? NS2 : m;
  }
  bitonic_desc(ssc, sid, NS2);
  const int vc2 = (int)scal[2];

  float* rois  = out;                                  // [B, maxbox, 4]
  float* scout = out + (size_t)B * maxbox * 4;         // [B, maxbox]
  int*   clout = (int*)(out + (size_t)B * maxbox * 5); // [B, maxbox]
  for (int r = tid; r < maxbox; r += BDIM) {
    bool val = (r < vc2);
    int fi = val ? sid[r] : 0;
    const float* bp = ws_boxes + ((size_t)b * N2 + fi) * 4;
    size_t o = (size_t)b * maxbox + r;
    rois[o * 4 + 0] = val ? bp[0] : 0.0f;
    rois[o * 4 + 1] = val ? bp[1] : 0.0f;
    rois[o * 4 + 2] = val ? bp[2] : 0.0f;
    rois[o * 4 + 3] = val ? bp[3] : 0.0f;
    scout[o] = val ? ssc[r] : 0.0f;
    clout[o] = val ? (fi / KCAP) : -1;
  }
}

// ===================== host launcher =====================
extern "C" void kernel_launch(void* const* d_in, const int* in_sizes, int n_in,
                              void* d_out, int out_size, void* d_ws, size_t ws_size,
                              hipStream_t stream) {
  (void)n_in; (void)ws_size;
  const float* cls = (const float*)d_in[3];   // classification [B,A,C]
  const float* ta  = (const float*)d_in[4];   // transformed_anchors [B,A,4]
  const float* thr = (const float*)d_in[5];   // threshold scalar

  const int A = in_sizes[1] / 4;              // anchors [1,A,4]
  const int B = in_sizes[2] / (4 * A);        // regression [B,A,4]
  const int C = in_sizes[3] / (A * B);        // classification [B,A,C]
  const int maxbox = out_size / (6 * B);      // rois(4)+scores(1)+class(1)

  float* ws_scores = (float*)d_ws;                         // [B*C, KCAP]
  float* ws_boxes  = ws_scores + (size_t)B * C * KCAP;     // [B*C, KCAP, 4]

  (void)hipFuncSetAttribute((const void*)pp_stage1,
                            hipFuncAttributeMaxDynamicSharedMemorySize, (int)S1_BYTES);
  (void)hipFuncSetAttribute((const void*)pp_stage2,
                            hipFuncAttributeMaxDynamicSharedMemorySize, (int)S2_BYTES);

  pp_stage1<<<B * C, BDIM, S1_BYTES, stream>>>(cls, ta, thr, B, A, C,
                                               ws_scores, ws_boxes);
  pp_stage2<<<B, BDIM, S2_BYTES, stream>>>(ws_scores, ws_boxes, thr, B, C, maxbox,
                                           (float*)d_out);
}
